// UMABackbone_70609262346619
// MI455X (gfx1250) — matrix operand order repaired and checked by reading, not compiled, not run on hardware
//
#include <hip/hip_runtime.h>
#include <math.h>

typedef _Float16 v16h __attribute__((ext_vector_type(16)));
typedef _Float16 v8h  __attribute__((ext_vector_type(8)));
typedef float    v8f  __attribute__((ext_vector_type(8)));

#define LMAX   2
#define SPH    9
#define CH     64
#define HID    128
#define NRBF   64
#define TILE_E 16
#define ROWS   (TILE_E * SPH)   /* 144 */
#define NODE_STRIDE (SPH * CH)  /* 576 floats = 2304 B */

// ---------------- dynamic LDS layout (bytes) ----------------
// NOTE: kernel uses ONLY dynamic LDS, so the group segment starts at LDS
// offset 0 and these byte offsets are directly usable as the VDST LDS
// address for global_load_async_to_lds_*.
#define OFF_BUFA   0              /* float[144*64]  gathered x, later z   36864 */
#define OFF_XROTA  36864          /* half [144*64]  GEMM1 A operand       18432 */
#define OFF_XROTB  55296          /* half [144*64]  GEMM2 A operand       18432 */
#define OFF_WM1T   73728          /* half [64*64]   W_msg1^T               8192 */
#define OFF_WM2T   81920          /* half [64*64]   W_msg2^T               8192 */
#define OFF_WR1T   90112          /* half [128*64]  W_rad1^T              16384 */
#define OFF_WR2T   106496         /* half [64*128]  W_rad2^T              16384 */
#define OFF_RBF    122880         /* half [16*64]                          2048 */
#define OFF_HID    124928         /* half [16*128]                         4096 */
#define OFF_RAD    129024         /* float[16*64]                          4096 */
#define OFF_WIG    133120         /* float[16*81]                          5184 */
#define OFF_J      138304         /* float[34] : J1[9] then J2[25]          136 */
#define OFF_SCAL   138440         /* dist/env/beta/alpha f32[16]*4 + src/dst int[16]*2 */
#define SMEM_BYTES 138880

#define WGT_BYTES  49152          /* 4 f16 weight blobs, contiguous in ws */
#define WGT_CHUNKS (WGT_BYTES / 16)      /* 3072 x b128 */
#define GATHER_CHUNKS (TILE_E * NODE_STRIDE * 4 / 16)  /* 2304 x b128 */

// ---------------- CDNA5 async copy global->LDS ----------------
__device__ inline void async_copy_b128(unsigned lds_byte_off, const void* gsrc) {
    asm volatile("global_load_async_to_lds_b128 %0, %1, off"
                 :: "v"(lds_byte_off), "v"(gsrc)
                 : "memory");
}
__device__ inline void wait_async0() {
    asm volatile("s_wait_asynccnt 0x0" ::: "memory");
}

// ---------------- WMMA helpers ----------------
__device__ inline v8f wmma_f16(v16h a, v16h b, v8f c) {
    return __builtin_amdgcn_wmma_f32_16x16x32_f16(false, a, false, b, (short)0, c, false, false);
}

// A operand (16x32 f16, MxK): lanes 0-15 row m hold K=kb+0..15,
// lanes 16-31 row m hold K=kb+16..31 (same k order as ldsB).
__device__ inline v16h ldsA(const _Float16* base, int row0, int strideH, int kb, int lane) {
    int m   = lane & 15;
    int sel = (lane >> 4) & 1;
    const _Float16* p = base + (row0 + m) * strideH + kb + sel * 16;
    v8h lo = *(const v8h*)p;
    v8h hi = *(const v8h*)(p + 8);
    v16h r;
#pragma unroll
    for (int i = 0; i < 8; ++i) { r[i] = lo[i]; r[i + 8] = hi[i]; }
    return r;
}

// B operand (32x16 f16, KxN) from W^T stored row-major [n][K]:
// lanes 0-15 col n hold K=kb+0..15, lanes 16-31 col n hold K=kb+16..31.
__device__ inline v16h ldsB(const _Float16* base, int n0, int strideH, int kb, int lane) {
    int n   = lane & 15;
    int sel = (lane >> 4) & 1;
    const _Float16* p = base + (n0 + n) * strideH + kb + sel * 16;
    return *(const v16h*)p;
}

// ---------------- small-matrix helpers (Wigner) ----------------
__device__ inline void zrot(float ang, int l, float* M) {
    int n = 2 * l + 1;
    for (int i = 0; i < n * n; ++i) M[i] = 0.f;
    for (int i = 0; i < n; ++i) {
        float fr = (float)(l - i);
        M[i * n + i] = cosf(fr * ang);
        M[i * n + (n - 1 - i)] = sinf(fr * ang);  // reference quirk: overwrites center
    }
}

__device__ inline void mm_small(const float* A, const float* B, float* C, int n) {
    for (int i = 0; i < n; ++i)
        for (int j = 0; j < n; ++j) {
            float s = 0.f;
            for (int k = 0; k < n; ++k) s += A[i * n + k] * B[k * n + j];
            C[i * n + j] = s;
        }
}

__device__ void compute_J(int l, float* J) {
    const float fact[6] = {1.f, 1.f, 2.f, 6.f, 24.f, 120.f};
    int n = 2 * l + 1;
    float d[25];
    const float cs = 0.7071067811865476f;  // cos(pi/4) == sin(pi/4)
    for (int mp = -l; mp <= l; ++mp)
        for (int m = -l; m <= l; ++m) {
            float val = 0.f;
            int k0 = (m - mp > 0) ? (m - mp) : 0;
            int k1 = (l + m < l - mp) ? (l + m) : (l - mp);
            for (int k = k0; k <= k1; ++k) {
                float sg  = ((mp - m + k) & 1) ? -1.f : 1.f;
                float num = sqrtf(fact[l + mp] * fact[l - mp] * fact[l + m] * fact[l - m]);
                float den = fact[l + m - k] * fact[k] * fact[mp - m + k] * fact[l - mp - k];
                val += sg * (num / den) * powf(cs, (float)(2 * l + m - mp - 2 * k))
                                        * powf(cs, (float)(mp - m + 2 * k));
            }
            d[(mp + l) * n + (m + l)] = val;
        }
    float Cr[25], Ci[25];
    for (int i = 0; i < n * n; ++i) { Cr[i] = 0.f; Ci[i] = 0.f; }
    const float is2 = 0.7071067811865476f;
    for (int m = -l; m <= l; ++m) {
        int r = m + l;
        if (m == 0) Cr[l * n + l] = 1.f;
        else if (m > 0) {
            Cr[r * n + (-m + l)] = is2;
            Cr[r * n + r] = ((m & 1) ? -1.f : 1.f) * is2;
        } else {
            Ci[r * n + r] = is2;
            Ci[r * n + (-m + l)] = -((m & 1) ? -1.f : 1.f) * is2;
        }
    }
    float Tr[25], Ti[25];
    for (int i = 0; i < n; ++i)
        for (int j = 0; j < n; ++j) {
            float tr = 0.f, ti = 0.f;
            for (int k = 0; k < n; ++k) {
                tr += d[i * n + k] * Cr[j * n + k];
                ti -= d[i * n + k] * Ci[j * n + k];
            }
            Tr[i * n + j] = tr; Ti[i * n + j] = ti;
        }
    for (int i = 0; i < n; ++i)
        for (int j = 0; j < n; ++j) {
            float s = 0.f;
            for (int k = 0; k < n; ++k)
                s += Cr[i * n + k] * Tr[k * n + j] - Ci[i * n + k] * Ti[k * n + j];
            J[i * n + j] = s;
        }
}

// ---------------- prep kernel: f16 transposed weights + J -> ws ----------------
__global__ __launch_bounds__(256) void prep_kernel(
    const float* __restrict__ Wr1, const float* __restrict__ Wr2,
    const float* __restrict__ Wm1, const float* __restrict__ Wm2,
    char* __restrict__ ws) {
    _Float16* wm1T = (_Float16*)ws;            // [64][64]
    _Float16* wm2T = wm1T + 4096;              // [64][64]
    _Float16* wr1T = wm2T + 4096;              // [128][64]
    _Float16* wr2T = wr1T + 8192;              // [64][128]
    float*    Jws  = (float*)(ws + WGT_BYTES); // [34]

    int t = blockIdx.x * 256 + threadIdx.x;
    int nt = gridDim.x * 256;
    for (int f = t; f < 4096; f += nt) {
        int d = f >> 6, c = f & 63;
        wm1T[f] = (_Float16)Wm1[c * 64 + d];
        wm2T[f] = (_Float16)Wm2[c * 64 + d];
    }
    for (int f = t; f < 8192; f += nt) {   // wr1T[h][k] = Wr1[k][h]
        int h = f >> 6, k = f & 63;
        wr1T[f] = (_Float16)Wr1[k * 128 + h];
    }
    for (int f = t; f < 8192; f += nt) {   // wr2T[d][h] = Wr2[h][d]
        int d = f >> 7, h = f & 127;
        wr2T[f] = (_Float16)Wr2[h * 64 + d];
    }
    if (t < 2) compute_J(t + 1, Jws + (t == 0 ? 0 : 9));
}

// ---------------- fused per-edge-tile kernel ----------------
__global__ __launch_bounds__(256) void edge_msg_kernel(
    const float* __restrict__ node_feats, const float* __restrict__ evec,
    const int* __restrict__ eidx, const float* __restrict__ br1,
    const char* __restrict__ wgt, float* __restrict__ out, int E) {

    extern __shared__ __align__(32) char smem[];
    float*    bufA  = (float*)(smem + OFF_BUFA);      // gathered x, later z
    _Float16* xrotA = (_Float16*)(smem + OFF_XROTA);
    _Float16* xrotB = (_Float16*)(smem + OFF_XROTB);
    _Float16* wm1T  = (_Float16*)(smem + OFF_WM1T);
    _Float16* wm2T  = (_Float16*)(smem + OFF_WM2T);
    _Float16* wr1T  = (_Float16*)(smem + OFF_WR1T);
    _Float16* wr2T  = (_Float16*)(smem + OFF_WR2T);
    _Float16* sRbf  = (_Float16*)(smem + OFF_RBF);
    _Float16* sHid  = (_Float16*)(smem + OFF_HID);
    float*    sRad  = (float*)(smem + OFF_RAD);
    float*    sWig  = (float*)(smem + OFF_WIG);
    float*    sJ    = (float*)(smem + OFF_J);         // J1[9] then J2[25]
    float*    sDist = (float*)(smem + OFF_SCAL);
    float*    sEnv  = sDist + 16;
    float*    sBeta = sDist + 32;
    float*    sAlpha= sDist + 48;
    int*      sSrc  = (int*)(sDist + 64);
    int*      sDst  = (int*)(sDist + 80);

    const int tid  = threadIdx.x;
    const int lane = tid & 31;
    const int wave = tid >> 5;
    const int bid  = blockIdx.x;

    // ---- P0: kick off async weight->LDS copies; edge scalars; J; zero wig ----
#pragma unroll
    for (int i = 0; i < WGT_CHUNKS / 256; ++i) {          // 12 x b128 per thread
        int c = tid + 256 * i;
        async_copy_b128((unsigned)(OFF_WM1T + c * 16), wgt + (size_t)c * 16);
    }
    if (tid < 34) sJ[tid] = ((const float*)(wgt + WGT_BYTES))[tid];
    for (int f = tid; f < 16 * 81; f += 256) sWig[f] = 0.f;
    if (tid < TILE_E) {
        int eg = bid * TILE_E + tid;
        float vx = 0.f, vy = 0.f, vz = 0.f;
        int src = 0, dst = 0;
        if (eg < E) {
            vx = evec[eg * 3 + 0]; vy = evec[eg * 3 + 1]; vz = evec[eg * 3 + 2];
            src = eidx[eg]; dst = eidx[E + eg];
        }
        float nrm = sqrtf(vx * vx + vy * vy + vz * vz);
        float inv = 1.f / fmaxf(nrm, 1e-7f);
        float y = fminf(fmaxf(vy * inv, -1.f), 1.f);
        float x = fminf(fmaxf(vx * inv, -1.f), 1.f);
        float z = fminf(fmaxf(vz * inv, -1.f), 1.f);
        sBeta[tid]  = acosf(fminf(fmaxf(y, -1.f + 1e-7f), 1.f - 1e-7f));
        sAlpha[tid] = atan2f(x, z);
        sDist[tid]  = nrm;
        float dd = nrm * (1.f / 6.f);
        float d5 = dd * dd * dd * dd * dd;
        float env = 1.f - 21.f * d5 + 35.f * d5 * dd - 15.f * d5 * dd * dd;
        env = (dd < 1.f) ? env : 0.f;
        sEnv[tid] = (eg < E) ? env : 0.f;
        sSrc[tid] = src; sDst[tid] = dst;
    }
    __syncthreads();   // scalars + J visible (weight asyncs still in flight)

    // ---- P1: async gather of node rows; RBF; Wigner blocks (overlap) ----
#pragma unroll
    for (int i = 0; i < GATHER_CHUNKS / 256; ++i) {       // 9 x b128 per thread
        int c = tid + 256 * i;
        int e = c / 144;                                   // 144 chunks per edge row
        int r = c - e * 144;
        const char* g = (const char*)node_feats + (size_t)sSrc[e] * (NODE_STRIDE * 4) + r * 16;
        async_copy_b128((unsigned)(OFF_BUFA + c * 16), g);
    }
    {
        const float step  = 6.0f / 63.0f;
        const float coeff = -0.5f / (step * step);
        for (int f = tid; f < TILE_E * NRBF; f += 256) {
            int e = f >> 6, k = f & 63;
            float dv = sDist[e] - step * (float)k;
            sRbf[f] = (_Float16)expf(coeff * dv * dv);
        }
    }
    if (tid < TILE_E * 3) {   // one (edge, l) Wigner block per thread
        int e = tid / 3, l = tid % 3;
        int n = 2 * l + 1;
        float Xa[25], Xb[25], Xc[25], T1[25], T2[25];
        zrot(-0.0f, l, Xa);            // alpha arg = -gamma, gamma == 0
        zrot(-sBeta[e], l, Xb);
        zrot(-sAlpha[e], l, Xc);
        const float one = 1.f;
        const float* J = (l == 0) ? &one : ((l == 1) ? sJ : (sJ + 9));
        mm_small(Xa, J, T1, n);
        mm_small(T1, Xb, T2, n);
        mm_small(T2, J, T1, n);
        mm_small(T1, Xc, T2, n);
        int st = l * l;
        for (int i = 0; i < n; ++i)
            for (int j = 0; j < n; ++j)
                sWig[e * 81 + (st + i) * 9 + (st + j)] = T2[i * n + j];
    }
    wait_async0();      // this wave's async LDS writes (weights + gather) done
    __syncthreads();    // ... and everyone else's

    // ---- P2: radial MLP GEMM1 (16x128 = 16x64 @ 64x128) + silu ----
    {
        v8f acc = {};
#pragma unroll
        for (int kb = 0; kb < 64; kb += 32) {
            v16h a = ldsA(sRbf, 0, 64, kb, lane);
            v16h b = ldsB(wr1T, 16 * wave, 64, kb, lane);
            acc = wmma_f16(a, b, acc);
        }
        int mB = (lane >= 16) ? 8 : 0;
        int h  = 16 * wave + (lane & 15);
        float bias = br1[h];
#pragma unroll
        for (int v = 0; v < 8; ++v) {
            float x = acc[v] + bias;
            float s = x / (1.f + expf(-x));
            sHid[(v + mB) * 128 + h] = (_Float16)s;
        }
    }
    __syncthreads();

    // ---- P3: radial MLP GEMM2 (16x64 = 16x128 @ 128x64) * env ----
    if (wave < 4) {
        v8f acc = {};
#pragma unroll
        for (int kb = 0; kb < 128; kb += 32) {
            v16h a = ldsA(sHid, 0, 128, kb, lane);
            v16h b = ldsB(wr2T, 16 * wave, 128, kb, lane);
            acc = wmma_f16(a, b, acc);
        }
        int mB = (lane >= 16) ? 8 : 0;
        int n  = 16 * wave + (lane & 15);
#pragma unroll
        for (int v = 0; v < 8; ++v) {
            int m = v + mB;
            sRad[m * 64 + n] = acc[v] * sEnv[m];
        }
    }
    __syncthreads();

    // ---- P4: forward rotation x_rot = wig @ x  -> f16 A buffer ----
    for (int f = tid; f < ROWS * CH; f += 256) {
        int e = f / NODE_STRIDE, r = f - e * NODE_STRIDE;
        int i = r >> 6, c = r & 63;
        const float* w = sWig + e * 81 + i * 9;
        const float* x = bufA + e * NODE_STRIDE + c;
        float acc = 0.f;
#pragma unroll
        for (int j = 0; j < 9; ++j) acc += w[j] * x[j * 64];
        xrotA[(e * 9 + i) * 64 + c] = (_Float16)acc;
    }
    __syncthreads();

    // ---- P5: message GEMM1: (144x64) @ W_msg1, scaled by rad -> f16 ----
    for (int t = wave; t < 36; t += 8) {
        int mt = t >> 2, nt = t & 3;
        v8f acc = {};
#pragma unroll
        for (int kb = 0; kb < 64; kb += 32) {
            v16h a = ldsA(xrotA, 16 * mt, 64, kb, lane);
            v16h b = ldsB(wm1T, 16 * nt, 64, kb, lane);
            acc = wmma_f16(a, b, acc);
        }
        int mB  = (lane >= 16) ? 8 : 0;
        int col = 16 * nt + (lane & 15);
#pragma unroll
        for (int v = 0; v < 8; ++v) {
            int row = 16 * mt + v + mB;
            int e = row / 9;
            xrotB[row * 64 + col] = (_Float16)(acc[v] * sRad[e * 64 + col]);
        }
    }
    __syncthreads();

    // ---- P6: message GEMM2: (144x64) @ W_msg2 -> f32 z ----
    for (int t = wave; t < 36; t += 8) {
        int mt = t >> 2, nt = t & 3;
        v8f acc = {};
#pragma unroll
        for (int kb = 0; kb < 64; kb += 32) {
            v16h a = ldsA(xrotB, 16 * mt, 64, kb, lane);
            v16h b = ldsB(wm2T, 16 * nt, 64, kb, lane);
            acc = wmma_f16(a, b, acc);
        }
        int mB  = (lane >= 16) ? 8 : 0;
        int col = 16 * nt + (lane & 15);
#pragma unroll
        for (int v = 0; v < 8; ++v) {
            int row = 16 * mt + v + mB;
            bufA[row * 64 + col] = acc[v];
        }
    }
    __syncthreads();

    // ---- P7: backward rotation (wig^T @ z) + scatter-add ----
    for (int pr = tid; pr < TILE_E * CH; pr += 256) {
        int e = pr >> 6, c = pr & 63;
        int dst = sDst[e];
        const float* w = sWig + e * 81;
        const float* z = bufA + e * NODE_STRIDE + c;
        float* o = out + (size_t)dst * NODE_STRIDE + c;
#pragma unroll
        for (int i = 0; i < 9; ++i) {
            float acc = 0.f;
#pragma unroll
            for (int j = 0; j < 9; ++j) acc += w[j * 9 + i] * z[j * 64];
            atomicAdd(o + i * 64, acc);
        }
    }
}

// ---------------- in-place RMS norm over the 9 sph components ----------------
__global__ __launch_bounds__(256) void rms_norm_kernel(float* __restrict__ out,
                                                       const float* __restrict__ gamma,
                                                       int nNodes) {
    int t = blockIdx.x * blockDim.x + threadIdx.x;
    if (t >= nNodes * CH) return;
    int n = t >> 6, c = t & 63;
    float* p = out + (size_t)n * NODE_STRIDE + c;
    float vals[9];
    float s = 0.f;
#pragma unroll
    for (int i = 0; i < 9; ++i) { vals[i] = p[i * 64]; s += vals[i] * vals[i]; }
    float g = gamma[c] * rsqrtf(s * (1.f / 9.f) + 1e-7f);
#pragma unroll
    for (int i = 0; i < 9; ++i) p[i * 64] = vals[i] * g;
}

extern "C" void kernel_launch(void* const* d_in, const int* in_sizes, int n_in,
                              void* d_out, int out_size, void* d_ws, size_t ws_size,
                              hipStream_t stream) {
    const float* node_feats = (const float*)d_in[0];
    const float* evec       = (const float*)d_in[1];
    const int*   eidx       = (const int*)d_in[2];
    const float* Wr1        = (const float*)d_in[3];
    const float* br1        = (const float*)d_in[4];
    const float* Wr2        = (const float*)d_in[5];
    const float* Wm1        = (const float*)d_in[6];
    const float* Wm2        = (const float*)d_in[7];
    const float* gamma      = (const float*)d_in[8];
    float* out = (float*)d_out;
    char*  ws  = (char*)d_ws;

    const int E      = in_sizes[2] / 2;
    const int nNodes = in_sizes[0] / NODE_STRIDE;

    hipMemsetAsync(out, 0, (size_t)out_size * sizeof(float), stream);

    prep_kernel<<<12, 256, 0, stream>>>(Wr1, Wr2, Wm1, Wm2, ws);

    const int blocks = (E + TILE_E - 1) / TILE_E;
    edge_msg_kernel<<<blocks, 256, SMEM_BYTES, stream>>>(
        node_feats, evec, eidx, br1, ws, out, E);

    const int nt = nNodes * CH;
    rms_norm_kernel<<<(nt + 255) / 256, 256, 0, stream>>>(out, gamma, nNodes);
}
